// RefinementHead_11484742550032
// MI455X (gfx1250) — matrix operand order repaired and checked by reading, not compiled, not run on hardware
//
#include <hip/hip_runtime.h>
#include <hip/hip_bf16.h>

typedef __attribute__((ext_vector_type(16))) _Float16 v16h;
typedef __attribute__((ext_vector_type(8)))  _Float16 v8h;
typedef __attribute__((ext_vector_type(8)))  float    v8f;

#define NPROP   2048
#define NPTS    256
#define H1D     64
#define H2D     128
#define H3D     256
#define THREADS 256
#define MIN_PTS 4

// ---- LDS layout (bytes) ----
// sW2t : H2D*H1D f16 = 16384   @ 0        \  contiguous 81920-byte blob,
// sW3t : H3D*H2D f16 = 65536   @ 16384    /  async-DMA'd from d_ws
// sH1  : NPTS*H1D f16 = 32768  @ 81920
// sH2  : NPTS*H2D f16 = 65536  @ 114688
// sW1  : 3*H1D f32    = 768    @ 180224
// sB1  : H1D f32      = 256    @ 180992
// sB2  : H2D f32      = 512    @ 181248
// sB3  : H3D f32      = 1024   @ 181760
// sFeat: H3D f32      = 1024   @ 182784
#define LDS_BYTES   183808
#define WBLOB_BYTES 81920   // f16 W2t + W3t in d_ws

// A fragment: row-major MxK matrix in LDS, row stride ldk (f16 elems).
// A 16x32 layout used here (contiguous k, not ISA table order): lane m = l&15, hi = l>>4;
//   element e -> k = k0 + e + 16*hi
__device__ inline v16h load_A(const _Float16* base, int ldk, int m0, int k0, int lane) {
    const int m  = m0 + (lane & 15);
    const int hi = lane >> 4;
    const v8h lo  = *(const v8h*)(base + m * ldk + k0 + 16 * hi);
    const v8h hiv = *(const v8h*)(base + m * ldk + k0 + 16 * hi + 8);
    return __builtin_shufflevector(lo, hiv, 0,1,2,3,4,5,6,7,8,9,10,11,12,13,14,15);
}

// B fragment: weights stored transposed [n][k], row stride ldk.
// B 32x16 layout used here (contiguous k, same as load_A): lane n = l&15, hi = l>>4; element e -> k = k0 + e + 16*hi
__device__ inline v16h load_B(const _Float16* baseT, int ldk, int n0, int k0, int lane) {
    const int n  = n0 + (lane & 15);
    const int hi = lane >> 4;
    const v8h* p = (const v8h*)(baseT + n * ldk + k0 + 16 * hi);
    return __builtin_shufflevector(p[0], p[1], 0,1,2,3,4,5,6,7,8,9,10,11,12,13,14,15);
}

// One-time weight prep: fp32 row-major (K,N) -> f16 transposed [n][k] in d_ws.
// ws[0 .. 8191]      = W2t (128 x 64)
// ws[8192 .. 40959]  = W3t (256 x 128)
__global__ __launch_bounds__(THREADS)
void prep_weights_kernel(const float* __restrict__ W2, const float* __restrict__ W3,
                         _Float16* __restrict__ ws) {
    const int i = blockIdx.x * THREADS + threadIdx.x;   // 0 .. 40959
    if (i < H2D * H1D) {
        const int n = i >> 6, k = i & 63;
        ws[i] = (_Float16)W2[k * H2D + n];
    } else {
        const int j = i - H2D * H1D;                    // 0 .. 32767
        const int n = j >> 7, k = j & 127;
        ws[H2D * H1D + j] = (_Float16)W3[k * H3D + n];
    }
}

__global__ __launch_bounds__(THREADS)
void refine_head_kernel(const float* __restrict__ points,
                        const int*   __restrict__ lengths,
                        const float* __restrict__ proposals,
                        const float* __restrict__ W1, const float* __restrict__ b1,
                        const float* __restrict__ W2, const float* __restrict__ b2,
                        const float* __restrict__ W3, const float* __restrict__ b3,
                        const float* __restrict__ Wc, const float* __restrict__ bc,
                        const float* __restrict__ Wr, const float* __restrict__ br,
                        const _Float16* __restrict__ wsW,   // pre-packed f16 weights (or null)
                        float* __restrict__ out)
{
    extern __shared__ char smem[];
    _Float16* sW2t  = (_Float16*)(smem);            // [H2D][H1D]  (transposed)
    _Float16* sW3t  = (_Float16*)(smem + 16384);    // [H3D][H2D]  (transposed)
    _Float16* sH1   = (_Float16*)(smem + 81920);    // [NPTS][H1D]
    _Float16* sH2   = (_Float16*)(smem + 114688);   // [NPTS][H2D]
    float*    sW1   = (float*)(smem + 180224);      // [3][H1D]
    float*    sB1   = (float*)(smem + 180992);
    float*    sB2   = (float*)(smem + 181248);
    float*    sB3   = (float*)(smem + 181760);
    float*    sFeat = (float*)(smem + 182784);

    const int tid  = threadIdx.x;
    const int lane = tid & 31;
    const int wave = tid >> 5;
    const int pid  = blockIdx.x;

    // ---- Stage f16 weight blob into LDS ----
    if (wsW) {
        // CDNA5 async copy: 20 x b128 per thread, ASYNCcnt-tracked.
        // Flat LDS pointer truncated to 32 bits == LDS byte offset (ISA 10.2).
        const unsigned ldsbase = (unsigned)(size_t)(void*)smem;
        const unsigned long long gbase = (unsigned long long)(const void*)wsW;
        #pragma unroll
        for (int i = 0; i < WBLOB_BYTES / (THREADS * 16); ++i) {   // 20 iters
            const unsigned off = (unsigned)tid * 16u + (unsigned)i * (THREADS * 16u);
            const unsigned laddr = ldsbase + off;
            const unsigned long long ga = gbase + off;
            asm volatile("global_load_async_to_lds_b128 %0, %1, off"
                         :: "v"(laddr), "v"(ga) : "memory");
        }
        asm volatile("s_wait_asynccnt 0x0" ::: "memory");
    } else {
        // Fallback: transpose+convert from fp32 weights in-kernel.
        for (int i = tid; i < H2D * H1D; i += THREADS) {
            const int n = i >> 6, k = i & 63;
            sW2t[i] = (_Float16)W2[k * H2D + n];
        }
        for (int i = tid; i < H3D * H2D; i += THREADS) {
            const int n = i >> 7, k = i & 127;
            sW3t[i] = (_Float16)W3[k * H3D + n];
        }
    }
    if (tid < 3 * H1D) sW1[tid] = W1[tid];
    if (tid < H1D)     sB1[tid] = b1[tid];
    if (tid < H2D)     sB2[tid] = b2[tid];
    sB3[tid] = b3[tid];
    __syncthreads();

    // ---- Layer 1 (K=3, scalar fp32): one point per thread, packed b128 stores ----
    {
        const int Lraw = lengths[pid];
        const int L    = Lraw > 0 ? Lraw : 1;
        const int src  = tid % L;
        const float cx = proposals[pid * 6 + 0];
        const float cy = proposals[pid * 6 + 1];
        const float cz = proposals[pid * 6 + 2];
        const float hx = proposals[pid * 6 + 3] * 0.5f + 1e-6f;
        const float hy = proposals[pid * 6 + 4] * 0.5f + 1e-6f;
        const float hz = proposals[pid * 6 + 5] * 0.5f + 1e-6f;
        const long base = ((long)pid * NPTS + src) * 3;
        const float lx = (points[base + 0] - cx) / hx;
        const float ly = (points[base + 1] - cy) / hy;
        const float lz = (points[base + 2] - cz) / hz;
        #pragma unroll
        for (int k8 = 0; k8 < H1D; k8 += 8) {
            v8h pack;
            #pragma unroll
            for (int j = 0; j < 8; ++j) {
                const int k = k8 + j;
                float v = fmaf(lx, sW1[k],
                          fmaf(ly, sW1[H1D + k],
                          fmaf(lz, sW1[2 * H1D + k], sB1[k])));
                pack[j] = (_Float16)fmaxf(v, 0.0f);
            }
            *(v8h*)(sH1 + tid * H1D + k8) = pack;
        }
    }
    __syncthreads();

    // ---- Layer 2: (256x64) @ (64x128) via WMMA; each wave owns one N-tile ----
    {
        const int n0 = wave * 16;
        const int nn = n0 + (lane & 15);
        const int hi = lane >> 4;
        const float bias = sB2[nn];
        for (int mtile = 0; mtile < 16; ++mtile) {
            v8f acc = {};
            #pragma unroll
            for (int kt = 0; kt < 2; ++kt) {
                v16h a = load_A(sH1,  H1D, mtile * 16, kt * 32, lane);
                v16h b = load_B(sW2t, H1D, n0,         kt * 32, lane);
                acc = __builtin_amdgcn_wmma_f32_16x16x32_f16(
                          false, a, false, b, (short)0, acc, false, false);
            }
            #pragma unroll
            for (int r = 0; r < 8; ++r) {
                const int m = mtile * 16 + r + 8 * hi;
                sH2[m * H2D + nn] = (_Float16)fmaxf(acc[r] + bias, 0.0f);
            }
        }
    }
    __syncthreads();

    // ---- Layer 3 + fused max-pool: (256x128) @ (128x256); 2 N-tiles per wave ----
    {
        const int hi = lane >> 4;
        #pragma unroll
        for (int t = 0; t < 2; ++t) {
            const int n0 = (wave * 2 + t) * 16;
            const int nn = n0 + (lane & 15);
            const float bias = sB3[nn];
            v8f vmax = {};   // ReLU floor: max(relu(x)) == max(0, max(x))
            for (int mtile = 0; mtile < 16; ++mtile) {
                v8f acc = {};
                #pragma unroll
                for (int kt = 0; kt < 4; ++kt) {
                    v16h a = load_A(sH2,  H2D, mtile * 16, kt * 32, lane);
                    v16h b = load_B(sW3t, H2D, n0,         kt * 32, lane);
                    acc = __builtin_amdgcn_wmma_f32_16x16x32_f16(
                              false, a, false, b, (short)0, acc, false, false);
                }
                #pragma unroll
                for (int r = 0; r < 8; ++r)
                    vmax[r] = fmaxf(vmax[r], acc[r] + bias);
            }
            float mx = vmax[0];
            #pragma unroll
            for (int r = 1; r < 8; ++r) mx = fmaxf(mx, vmax[r]);
            mx = fmaxf(mx, __shfl_xor(mx, 16, 32));   // combine M halves (same n)
            if (lane < 16) sFeat[nn] = mx;
        }
    }
    __syncthreads();

    // ---- Heads: feats(256) -> cls(1), reg(4); one wave, shuffle reduction ----
    if (wave == 0) {
        const bool valid = lengths[pid] >= MIN_PTS;
        float acc0 = 0.f, acc1 = 0.f, acc2 = 0.f, acc3 = 0.f, acc4 = 0.f;
        for (int c = lane; c < H3D; c += 32) {
            const float f = valid ? sFeat[c] : 0.0f;
            acc0 = fmaf(f, Wc[c],         acc0);
            acc1 = fmaf(f, Wr[c * 4 + 0], acc1);
            acc2 = fmaf(f, Wr[c * 4 + 1], acc2);
            acc3 = fmaf(f, Wr[c * 4 + 2], acc3);
            acc4 = fmaf(f, Wr[c * 4 + 3], acc4);
        }
        #pragma unroll
        for (int off = 16; off >= 1; off >>= 1) {
            acc0 += __shfl_xor(acc0, off, 32);
            acc1 += __shfl_xor(acc1, off, 32);
            acc2 += __shfl_xor(acc2, off, 32);
            acc3 += __shfl_xor(acc3, off, 32);
            acc4 += __shfl_xor(acc4, off, 32);
        }
        if (lane == 0) {
            out[pid]                 = acc0 + bc[0];
            out[NPROP + pid * 4 + 0] = acc1 + br[0];
            out[NPROP + pid * 4 + 1] = acc2 + br[1];
            out[NPROP + pid * 4 + 2] = acc3 + br[2];
            out[NPROP + pid * 4 + 3] = acc4 + br[3];
        }
    }
}

extern "C" void kernel_launch(void* const* d_in, const int* in_sizes, int n_in,
                              void* d_out, int out_size, void* d_ws, size_t ws_size,
                              hipStream_t stream) {
    const float* points    = (const float*)d_in[0];
    const int*   lengths   = (const int*)  d_in[1];
    const float* proposals = (const float*)d_in[2];
    const float* W1 = (const float*)d_in[3];  const float* b1 = (const float*)d_in[4];
    const float* W2 = (const float*)d_in[5];  const float* b2 = (const float*)d_in[6];
    const float* W3 = (const float*)d_in[7];  const float* b3 = (const float*)d_in[8];
    const float* Wc = (const float*)d_in[9];  const float* bc = (const float*)d_in[10];
    const float* Wr = (const float*)d_in[11]; const float* br = (const float*)d_in[12];
    float* out = (float*)d_out;

    _Float16* wsW = nullptr;
    if (ws_size >= (size_t)WBLOB_BYTES) {
        wsW = (_Float16*)d_ws;
        // (H2D*H1D + H3D*H2D) / THREADS = 160 blocks, exact.
        prep_weights_kernel<<<(H2D * H1D + H3D * H2D) / THREADS, THREADS, 0, stream>>>(
            W2, W3, wsW);
    }

    // Dynamic LDS > 64KB: raise the opt-in cap (host-side attribute, capture-safe).
    (void)hipFuncSetAttribute((const void*)refine_head_kernel,
                              hipFuncAttributeMaxDynamicSharedMemorySize, LDS_BYTES);

    refine_head_kernel<<<NPROP, THREADS, LDS_BYTES, stream>>>(
        points, lengths, proposals, W1, b1, W2, b2, W3, b3, Wc, bc, Wr, br, wsW, out);
}
